// LearnedEdge_16896401342533
// MI455X (gfx1250) — matrix operand order repaired and checked by reading, not compiled, not run on hardware
//
#include <hip/hip_runtime.h>

#define FEAT     128
#define NNODES   512
#define EDGES    97920
#define NSAMP    5
#define NBATCH   4
#define TRI_OFF  32896   // 256*257/2 : edges skipped before row 257

// ---- clang ext-vector types ----
typedef __attribute__((ext_vector_type(16))) __bf16        v16bf;
typedef __attribute__((ext_vector_type(8)))  float         v8f;
typedef __attribute__((ext_vector_type(4)))  float         f32x4;
typedef __attribute__((ext_vector_type(4)))  unsigned int  u32x4;
typedef __attribute__((ext_vector_type(2)))  unsigned int  u32x2;
typedef __attribute__((ext_vector_type(8)))  int           i32x8;
typedef __attribute__((ext_vector_type(4)))  int           i32x4;

struct Frag16 { u32x4 q0, q1; };   // 32 B == one v16bf WMMA operand

__device__ __forceinline__ v16bf frag_cast(const Frag16& f) {
  return __builtin_bit_cast(v16bf, f);
}

// f32 -> bf16 round-to-nearest-even
__device__ __forceinline__ unsigned short f2bf(float f) {
  unsigned int u = __float_as_uint(f);
  unsigned int r = u + 0x7FFFu + ((u >> 16) & 1u);
  return (unsigned short)(r >> 16);
}
__device__ __forceinline__ unsigned int pack2bf(float a, float b) {
  return (unsigned int)f2bf(a) | ((unsigned int)f2bf(b) << 16);
}

// flat edge index -> (sink row r, source col c): r>=257, c<r
__device__ __forceinline__ void edge_rc(int e, int* r, int* c) {
  long q = (long)e + TRI_OFF;
  int rr = (int)((1.0 + sqrt(1.0 + 8.0 * (double)q)) * 0.5);
  while ((long)rr * (rr - 1) / 2 > q) --rr;
  while ((long)(rr + 1) * rr / 2 <= q) ++rr;
  *r = rr;
  *c = (int)(q - (long)rr * (rr - 1) / 2);
}

// deterministic per-(b,s,e) gumbel noise (counter-based hash)
__device__ __forceinline__ float gumbelf(int b, int s, int e) {
  unsigned int x = (unsigned int)e * 2654435761u ^
                   ((unsigned int)(b * NSAMP + s + 1) * 0x9E3779B9u);
  x ^= x >> 16; x *= 0x7feb352du;
  x ^= x >> 15; x *= 0x846ca68bu;
  x ^= x >> 16;
  float u = (float)(x >> 8) * (1.0f / 16777216.0f) + 5.9604645e-08f;
  return -__logf(-__logf(u));
}

// ------------------------------------------------------------------
// Tensor Data Mover helpers (D# built per CDNA5 ISA)
#if __has_include(<hip/amd_detail/amd_gfx1250_TDM.h>)
#define TDM_ARITY6 1
#endif

__device__ __forceinline__ void tdm_issue(u32x4 g0, i32x8 g1, i32x4 g2, i32x4 g3) {
#ifdef TDM_ARITY6
  i32x8 gx = {};
  __builtin_amdgcn_tensor_load_to_lds(g0, g1, g2, g3, gx, 0);
#else
  __builtin_amdgcn_tensor_load_to_lds(g0, g1, g2, g3, 0);
#endif
}

// D# group0: count=1, lds_addr, 57b global addr, type=2 ("image"), optional gather
__device__ __forceinline__ u32x4 tdm_g0(const void* gp, unsigned ldsOff, bool gather) {
  unsigned long long ga = (unsigned long long)gp;
  u32x4 g0;
  g0.x = 1u | (gather ? (1u << 31) : 0u);   // gather_index_size=0 (16-bit)
  g0.y = ldsOff;
  g0.z = (unsigned int)ga;
  g0.w = (unsigned int)((ga >> 32) & 0x01FFFFFFu) | (2u << 30);
  return g0;
}

// D# group1: data size, pad config, tensor_dim0/1, tile_dim0/1, dim0 stride
__device__ __forceinline__ i32x8 tdm_g1(unsigned dszCode, unsigned rowElems,
                                        unsigned tdim1, unsigned tile1,
                                        unsigned strideElems,
                                        unsigned padInt, unsigned padAmt) {
  i32x8 g;
  g[0] = (int)((dszCode << 16) | (1u << 20) | (padInt << 22) | (padAmt << 25));
  g[1] = (int)((rowElems & 0xFFFFu) << 16);                                // t_dim0 lo
  g[2] = (int)(((rowElems >> 16) & 0xFFFFu) | ((tdim1 & 0xFFFFu) << 16));  // t_dim0 hi | t_dim1 lo
  g[3] = (int)(((tdim1 >> 16) & 0xFFFFu) | ((rowElems & 0xFFFFu) << 16));  // t_dim1 hi | tile_dim0
  g[4] = (int)(tile1 & 0xFFFFu);                                           // tile_dim1 (tile_dim2=0)
  g[5] = (int)strideElems;                                                 // dim0_stride lo32
  g[6] = 0;
  g[7] = 0;
  return g;
}

// ------------------------------------------------------------------
// prep: W1 -> bf16 [n][k] transposed, W2 -> bf16 [n][k], nodes -> bf16
__global__ void prep_kernel(const float* __restrict__ nodes,
                            const float* __restrict__ W1,
                            const float* __restrict__ W2,
                            unsigned short* __restrict__ W1t,
                            unsigned short* __restrict__ W2t,
                            unsigned short* __restrict__ nodesBf) {
  int i = blockIdx.x * blockDim.x + threadIdx.x;
  if (i < 32768) {
    int k = i >> 7, n = i & 127;
    W1t[n * 256 + k] = f2bf(W1[i]);
  } else if (i < 49152) {
    int j = i - 32768, k = j >> 7, n = j & 127;
    W2t[n * 128 + k] = f2bf(W2[j]);
  } else if (i < 49152 + NBATCH * NNODES * FEAT) {
    int j = i - 49152;
    nodesBf[j] = f2bf(nodes[j]);
  }
}

// zero adjacency output + argmax slots
__global__ void init_kernel(float* __restrict__ out,
                            unsigned long long* __restrict__ slots) {
  long i = (long)blockIdx.x * blockDim.x + threadIdx.x;
  if (i < (long)NBATCH * NNODES * NNODES) out[i] = 0.0f;
  if (i < NBATCH * NSAMP) slots[i] = 0ull;
}

// ------------------------------------------------------------------
// LDS layout (bank-conflict-free padded rows; pad inserted by TDM):
//   sW1 @0      : 128 rows x 264 bf16 (528B)  = 67584 B
//   sW2 @67584  : 128 rows x 136 bf16 (272B)  = 34816 B
//   sA  @102400 : per wave 16 x 264 bf16      =  8448 B x 16  (overlay sH f32 [16][132])
//   sHb @237568 : per wave 16 x 136 bf16      =  4352 B x 16
// total 307200 B <= 320KB/WGP
#define SA_BASE  102400u
#define SHB_BASE 237568u

__global__ __launch_bounds__(512, 1) void mlp_kernel(
    const unsigned short* __restrict__ nodesBf,  // [B][512][128] bf16
    const float* __restrict__ bias1, const float* __restrict__ gam1, const float* __restrict__ bet1,
    const float* __restrict__ bias2, const float* __restrict__ gam2, const float* __restrict__ bet2,
    const float* __restrict__ W3, const float* __restrict__ b3,
    const unsigned short* __restrict__ Wb,       // W1t(65536B) ++ W2t(32768B)
    unsigned long long* __restrict__ slots) {
  extern __shared__ char smem[];
  unsigned short* sW1 = (unsigned short*)smem;
  unsigned short* sW2 = (unsigned short*)(smem + 67584);

  const int tid  = threadIdx.x;
  const int wave = tid >> 5;
  const int lane = tid & 31;

  unsigned short* sA  = (unsigned short*)(smem + SA_BASE + wave * 8448);
  float*          sH  = (float*)(smem + SA_BASE + wave * 8448);   // overlay
  unsigned short* sHb = (unsigned short*)(smem + SHB_BASE + wave * 4352);

  const int b        = blockIdx.y;
  const int tileBase = (blockIdx.x * 16 + wave) * 16;
  const bool active  = tileBase < EDGES;

  // ---- TDM: weights -> LDS (with pad-on-the-fly), one wave issues ----
  if (tid == 0) {
    // W1t: 128 rows x 128 dw, +4dw pad every 128 dw -> 528B LDS rows
    tdm_issue(tdm_g0(Wb, 0u, false),
              tdm_g1(2u, 128u, 128u, 128u, 128u, 6u, 3u), i32x4{}, i32x4{});
    // W2t: 128 rows x 64 dw, +4dw pad every 64 dw -> 272B LDS rows
    tdm_issue(tdm_g0(Wb + 32768, 67584u, false),
              tdm_g1(2u, 64u, 128u, 128u, 64u, 5u, 3u), i32x4{}, i32x4{});
  }

  // ---- TDM gather mode: 16 sink rows + 16 src rows -> per-wave A tile ----
  if (active) {
    unsigned short sidx[16], cidx[16];
    int r, c;
    edge_rc(tileBase, &r, &c);
#pragma unroll
    for (int m = 0; m < 16; ++m) {
      sidx[m] = (unsigned short)r;
      cidx[m] = (unsigned short)c;
      if (++c == r) { c = 0; ++r; }
    }
    i32x4 g2a, g3a, g2b, g3b;
#pragma unroll
    for (int j = 0; j < 4; ++j) {
      g2a[j] = (int)((unsigned)sidx[2 * j]     | ((unsigned)sidx[2 * j + 1] << 16));
      g3a[j] = (int)((unsigned)sidx[8 + 2 * j] | ((unsigned)sidx[9 + 2 * j] << 16));
      g2b[j] = (int)((unsigned)cidx[2 * j]     | ((unsigned)cidx[2 * j + 1] << 16));
      g3b[j] = (int)((unsigned)cidx[8 + 2 * j] | ((unsigned)cidx[9 + 2 * j] << 16));
    }
    const unsigned short* nbase = nodesBf + (long)b * NNODES * FEAT;
    const unsigned saOff = SA_BASE + (unsigned)wave * 8448u;
    // each row: 128 bf16 (64 dw) data, then 68 dw LDS pad -> 528B row stride
    tdm_issue(tdm_g0(nbase, saOff, true),
              tdm_g1(1u, 128u, NNODES, 16u, 128u, 5u, 67u), g2a, g3a);
    tdm_issue(tdm_g0(nbase, saOff + 256u, true),
              tdm_g1(1u, 128u, NNODES, 16u, 128u, 5u, 67u), g2b, g3b);
  }

  __builtin_amdgcn_s_wait_tensorcnt((short)0);  // own wave's TDM ops done
  __syncthreads();                              // wave0's weight loads visible

  const int hl    = lane & 15;
  const int koff  = (lane >> 4) * 8;    // A-frag K sub-offset (lane halves)
  const int bko   = (lane >> 4) * 8;    // B-frag K sub-offset (lane halves)
  const int mbase = (lane >> 4) * 8;    // C-frag row base

  // ================= layer 1: [16x256] @ [256x128] =================
  v8f acc[8] = {};
#pragma unroll
  for (int kb = 0; kb < 8; ++kb) {
    const int K0 = kb * 32;
    Frag16 fa;
    fa.q0 = *(const u32x4*)(sA + hl * 264 + K0 + koff);
    fa.q1 = *(const u32x4*)(sA + hl * 264 + K0 + 16 + koff);
    const v16bf av = frag_cast(fa);
#pragma unroll
    for (int nb = 0; nb < 8; ++nb) {
      Frag16 fb;
      const unsigned short* wp = sW1 + (nb * 16 + hl) * 264 + K0 + bko;
      fb.q0 = *(const u32x4*)(wp);
      fb.q1 = *(const u32x4*)(wp + 16);
      acc[nb] = __builtin_amdgcn_wmma_f32_16x16x32_bf16(
          false, av, false, frag_cast(fb), (short)0, acc[nb], false, false);
    }
  }

  // bias + relu -> per-wave f32 scratch (overlays finished A tile, stride 132)
  __builtin_amdgcn_wave_barrier();
#pragma unroll
  for (int nb = 0; nb < 8; ++nb) {
    const int n = nb * 16 + hl;
    const float bias = bias1[n];
#pragma unroll
    for (int i = 0; i < 8; ++i) {
      float v = acc[nb][i] + bias;
      sH[(mbase + i) * 132 + n] = v > 0.0f ? v : 0.0f;
    }
  }
  __builtin_amdgcn_wave_barrier();

  // layernorm 1 -> bf16 activations (stride 136)
  if (lane < 16) {
    const float* row = sH + lane * 132;
    f32x4 s4 = {}, q4 = {};
    for (int n = 0; n < 128; n += 4) {
      f32x4 v = *(const f32x4*)(row + n);
      s4 += v; q4 += v * v;
    }
    const float s  = s4.x + s4.y + s4.z + s4.w;
    const float ss = q4.x + q4.y + q4.z + q4.w;
    const float mu  = s * (1.0f / 128.0f);
    const float inv = rsqrtf(ss * (1.0f / 128.0f) - mu * mu + 1e-5f);
    unsigned short* orow = sHb + lane * 136;
    for (int n = 0; n < 128; n += 4) {
      f32x4 v = *(const f32x4*)(row + n);
      f32x4 gg = *(const f32x4*)(gam1 + n);
      f32x4 bb = *(const f32x4*)(bet1 + n);
      f32x4 o = (v - mu) * inv * gg + bb;
      u32x2 p = { pack2bf(o.x, o.y), pack2bf(o.z, o.w) };
      *(u32x2*)(orow + n) = p;
    }
  }
  __builtin_amdgcn_wave_barrier();

  // ================= layer 2: [16x128] @ [128x128] =================
  v8f acc2[8] = {};
#pragma unroll
  for (int kb = 0; kb < 4; ++kb) {
    const int K0 = kb * 32;
    Frag16 fa;
    fa.q0 = *(const u32x4*)(sHb + hl * 136 + K0 + koff);
    fa.q1 = *(const u32x4*)(sHb + hl * 136 + K0 + 16 + koff);
    const v16bf av = frag_cast(fa);
#pragma unroll
    for (int nb = 0; nb < 8; ++nb) {
      Frag16 fb;
      const unsigned short* wp = sW2 + (nb * 16 + hl) * 136 + K0 + bko;
      fb.q0 = *(const u32x4*)(wp);
      fb.q1 = *(const u32x4*)(wp + 16);
      acc2[nb] = __builtin_amdgcn_wmma_f32_16x16x32_bf16(
          false, av, false, frag_cast(fb), (short)0, acc2[nb], false, false);
    }
  }

  __builtin_amdgcn_wave_barrier();
#pragma unroll
  for (int nb = 0; nb < 8; ++nb) {
    const int n = nb * 16 + hl;
    const float bias = bias2[n];
#pragma unroll
    for (int i = 0; i < 8; ++i) {
      float v = acc2[nb][i] + bias;
      sH[(mbase + i) * 132 + n] = v > 0.0f ? v : 0.0f;
    }
  }
  __builtin_amdgcn_wave_barrier();

  // layernorm 2 + W3 dot + straight-through gumbel argmax (fused)
  if (active && lane < 16) {
    const float* row = sH + lane * 132;
    f32x4 s4 = {}, q4 = {};
    for (int n = 0; n < 128; n += 4) {
      f32x4 v = *(const f32x4*)(row + n);
      s4 += v; q4 += v * v;
    }
    const float s  = s4.x + s4.y + s4.z + s4.w;
    const float ss = q4.x + q4.y + q4.z + q4.w;
    const float mu  = s * (1.0f / 128.0f);
    const float inv = rsqrtf(ss * (1.0f / 128.0f) - mu * mu + 1e-5f);
    f32x4 d4 = {};
    for (int n = 0; n < 128; n += 4) {
      f32x4 v = *(const f32x4*)(row + n);
      f32x4 gg = *(const f32x4*)(gam2 + n);
      f32x4 bb = *(const f32x4*)(bet2 + n);
      f32x4 w4 = *(const f32x4*)(W3 + n);
      d4 += ((v - mu) * inv * gg + bb) * w4;
    }
    const float logit = d4.x + d4.y + d4.z + d4.w + b3[0];
    const int e = tileBase + lane;
#pragma unroll
    for (int s5 = 0; s5 < NSAMP; ++s5) {
      const float val = logit + gumbelf(b, s5, e);
      unsigned int u = __float_as_uint(val);
      u = (u & 0x80000000u) ? ~u : (u | 0x80000000u);  // order-preserving key
      const unsigned long long key =
          ((unsigned long long)u << 32) | (unsigned int)e;
      atomicMax(&slots[b * NSAMP + s5], key);
    }
  }
}

// write 1.0 at each (b, sample)-argmax edge
__global__ void scatter_kernel(const unsigned long long* __restrict__ slots,
                               float* __restrict__ out) {
  int i = threadIdx.x;
  if (i < NBATCH * NSAMP) {
    unsigned long long k = slots[i];
    int e = (int)(k & 0xFFFFFFFFull);
    int b = i / NSAMP;
    int r, c;
    edge_rc(e, &r, &c);
    out[((long)b * NNODES + r) * NNODES + c] = 1.0f;
  }
}

// ------------------------------------------------------------------
extern "C" void kernel_launch(void* const* d_in, const int* in_sizes, int n_in,
                              void* d_out, int out_size, void* d_ws, size_t ws_size,
                              hipStream_t stream) {
  (void)in_sizes; (void)n_in; (void)out_size; (void)ws_size;
  const float* nodes = (const float*)d_in[0];
  const float* W1  = (const float*)d_in[1];
  const float* b1  = (const float*)d_in[2];
  const float* g1  = (const float*)d_in[3];
  const float* be1 = (const float*)d_in[4];
  const float* W2  = (const float*)d_in[5];
  const float* b2  = (const float*)d_in[6];
  const float* g2  = (const float*)d_in[7];
  const float* be2 = (const float*)d_in[8];
  const float* W3  = (const float*)d_in[9];
  const float* b3  = (const float*)d_in[10];
  float* out = (float*)d_out;

  unsigned short* W1t     = (unsigned short*)d_ws;                    // 65536 B
  unsigned short* W2t     = W1t + 256 * 128;                          // 32768 B
  unsigned short* nodesBf = (unsigned short*)((char*)d_ws + 98304);   // 524288 B
  unsigned long long* slots =
      (unsigned long long*)((char*)d_ws + 622592);                    // 160 B

  prep_kernel<<<1216, 256, 0, stream>>>(nodes, W1, W2, W1t, W2t, nodesBf);
  init_kernel<<<4096, 256, 0, stream>>>(out, slots);

  dim3 grid(383, NBATCH);  // 383 blocks x 16 waves x 16 edges >= 97920
  mlp_kernel<<<grid, 512, 307200, stream>>>(nodesBf, b1, g1, be1, b2, g2, be2,
                                            W3, b3, W1t, slots);
  scatter_kernel<<<1, 32, 0, stream>>>(slots, out);
}
